// StackRNN_44564580663298
// MI455X (gfx1250) — matrix operand order repaired and checked by reading, not compiled, not run on hardware
//
#include <hip/hip_runtime.h>
#include <hip/hip_bf16.h>

// ---------------------------------------------------------------------------
// StackRNN forward for MI455X (gfx1250, wave32, WMMA).
//
// Sizes: VOC=16000 EDIM=HDIM=128 SEQ=128 BSZ=32 N=16 NACT=18, H2=64.
// Roofline: logits output = 2x127x32x16000 fp32 = 520MB stores -> ~22us at
// 23.3TB/s; dedup'd GEMM is only 16.8 GFLOP, so f16 WMMA keeps it HBM-bound
// (fp32 WMMA 16x16x4 would make it compute-bound). emb_w-f16 (4MB) and
// buf_outs-f16 (1MB) stay L2-resident; outputs use non-temporal stores.
// Sequential parts: GRU recurrence via per-step WMMA (W_hh fragments pinned
// in registers), stack automaton split into 32 independent per-batch scans.
// Fast v_rcp_f32-based sigmoid/tanh shortens the dependent chains in the
// latency-bound scan kernels.
// ---------------------------------------------------------------------------

typedef _Float16 half16 __attribute__((ext_vector_type(16)));
typedef float    float8 __attribute__((ext_vector_type(8)));

#define SEQ   128
#define BSZ   32
#define EDIM  128
#define HDIM  128
#define H2    64
#define VOC   16000
#define NSTK  16
#define NACT  18
#define ROWS  (SEQ*BSZ)              // 4096
#define LOGROWS (127*BSZ)            // 4064
#define RIGHT_OFF ((size_t)127*BSZ*VOC)     // 65,024,000
#define NEG_OFF   ((size_t)2*127*BSZ*VOC)   // 130,048,000

// Fast transcendentals: v_rcp_f32 instead of IEEE div sequence (latency-bound
// sequential kernels care about dependent-chain length, not last-ulp).
__device__ __forceinline__ float fast_rcp(float x)     { return __builtin_amdgcn_rcpf(x); }
__device__ __forceinline__ float sigmoidf_(float x)    { return fast_rcp(1.0f + __expf(-x)); }
__device__ __forceinline__ float tanhf_(float x) {
    // tanh(x) = 1 - 2/(exp(2x)+1), stable for large |x| via clamping exp arg.
    float e = __expf(fminf(fmaxf(2.0f * x, -40.0f), 40.0f));
    return 1.0f - 2.0f * fast_rcp(e + 1.0f);
}

// ---------------------------------------------------------------------------
// K0: embedding gather (PAD==0 -> zero row)
__global__ void embed_gather(const int* __restrict__ inp, const float* __restrict__ emb_w,
                             float* __restrict__ embs) {
    int r = blockIdx.x;           // s*32+b
    int h = threadIdx.x;          // 0..127
    int tok = inp[r];
    float v = (tok == 0) ? 0.0f : emb_w[(size_t)tok * EDIM + h];
    embs[(size_t)r * EDIM + h] = v;
}

// K0b: fp32 -> f16 cast
__global__ void cast_f16(const float* __restrict__ src, _Float16* __restrict__ dst, int n) {
    int i = blockIdx.x * blockDim.x + threadIdx.x;
    if (i < n) dst[i] = (_Float16)src[i];
}

// ---------------------------------------------------------------------------
// K1: gi = embs @ [bwf_ih | bwb_ih | cw_ih]^T + bias   -> GI (4096 x 768)
__global__ void gi_precompute(const float* __restrict__ embs,
                              const float* __restrict__ bwf_ih, const float* __restrict__ bbf_ih,
                              const float* __restrict__ bwb_ih, const float* __restrict__ bbb_ih,
                              const float* __restrict__ cw_ih,  const float* __restrict__ cb_ih,
                              float* __restrict__ GI) {
    int idx = blockIdx.x * 256 + threadIdx.x;
    if (idx >= ROWS * 768) return;
    int r = idx / 768, c = idx - r * 768;
    const float* w; float bias;
    if (c < 192)      { w = bwf_ih + (size_t)c * EDIM;        bias = bbf_ih[c]; }
    else if (c < 384) { w = bwb_ih + (size_t)(c-192) * EDIM;  bias = bbb_ih[c-192]; }
    else              { w = cw_ih  + (size_t)(c-384) * EDIM;  bias = cb_ih[c-384]; }
    const float* x = embs + (size_t)r * EDIM;
    float acc = bias;
    #pragma unroll 8
    for (int k = 0; k < EDIM; ++k) acc += x[k] * w[k];
    GI[idx] = acc;
}

// ---------------------------------------------------------------------------
// K2: GRU scans (block 0 = fwd, block 1 = bwd), 256 threads = 8 waves.
// Recurrence gh = h(32x64) @ w_hh^T(64x192) done with WMMA f16 each step.
__global__ __launch_bounds__(256) void gru_scan(
        const float* __restrict__ GI,
        const float* __restrict__ whh_f, const float* __restrict__ bhh_f,
        const float* __restrict__ whh_b, const float* __restrict__ bhh_b,
        float* __restrict__ buf_outs) {
    const bool bwd = (blockIdx.x == 1);
    const float* whh = bwd ? whh_b : whh_f;
    const float* bhh = bwd ? bhh_b : bhh_f;
    const int giOff  = bwd ? 192 : 0;
    const int outOff = bwd ? H2 : 0;

    __shared__ float hlds[BSZ * H2];        // 8 KB
    __shared__ float ghlds[BSZ * 192];      // 24 KB

    const int tid  = threadIdx.x;
    const int lane = tid & 31;
    const int wave = tid >> 5;               // 0..7 ; 24 tiles -> 3 per wave
    const int lmod = lane & 15;
    const int lhi  = lane >> 4;

    // Preload B fragments (w_hh^T columns are w_hh rows, contiguous fp32).
    half16 bf[3][2];
    #pragma unroll
    for (int u = 0; u < 3; ++u) {
        int tIdx = wave * 3 + u;
        int ct = tIdx >> 1;                  // column tile 0..11
        int n  = ct * 16 + lmod;             // gate index 0..191
        #pragma unroll
        for (int ks = 0; ks < 2; ++ks) {
            int kb = ks * 32 + lhi * 8;
            #pragma unroll
            for (int i = 0; i < 16; ++i)
                bf[u][ks][i] = (_Float16)whh[(size_t)n * H2 + kb + (i & 7) + ((i >> 3) << 4)];
        }
    }
    for (int i = tid; i < BSZ * H2; i += 256) hlds[i] = 0.0f;
    __syncthreads();

    for (int t = 0; t < SEQ; ++t) {
        int s = bwd ? (SEQ - 1 - t) : t;
        // --- WMMA phase: gh = h @ w_hh^T ---
        #pragma unroll
        for (int u = 0; u < 3; ++u) {
            int tIdx = wave * 3 + u;
            int rt = tIdx & 1, ct = tIdx >> 1;
            float8 acc = {0.f,0.f,0.f,0.f,0.f,0.f,0.f,0.f};
            #pragma unroll
            for (int ks = 0; ks < 2; ++ks) {
                half16 af;
                int m  = rt * 16 + lmod;
                int kb = ks * 32 + lhi * 8;
                #pragma unroll
                for (int i = 0; i < 8; ++i) {
                    af[i]     = (_Float16)hlds[m * H2 + kb + i];
                    af[8 + i] = (_Float16)hlds[m * H2 + kb + 16 + i];
                }
                acc = __builtin_amdgcn_wmma_f32_16x16x32_f16(
                        false, af, false, bf[u][ks], (short)0, acc, false, false);
            }
            int mb = rt * 16 + (lhi << 3);
            int n  = ct * 16 + lmod;
            #pragma unroll
            for (int v = 0; v < 8; ++v) ghlds[(mb + v) * 192 + n] = acc[v];
        }
        __syncthreads();
        // --- gate phase: 2048 hidden units, 8 per thread ---
        #pragma unroll
        for (int u = 0; u < 8; ++u) {
            int idx = u * 256 + tid;
            int b = idx >> 6, j = idx & 63;
            int r = s * BSZ + b;
            const float* g = GI + (size_t)r * 768 + giOff;
            float ir = g[j], iz = g[64 + j], inn = g[128 + j];
            float ghr = ghlds[b * 192 + j]        + bhh[j];
            float ghz = ghlds[b * 192 + 64 + j]   + bhh[64 + j];
            float ghn = ghlds[b * 192 + 128 + j]  + bhh[128 + j];
            float h  = hlds[b * H2 + j];
            float rg = sigmoidf_(ir + ghr);
            float z  = sigmoidf_(iz + ghz);
            float nn = tanhf_(inn + rg * ghn);
            float hn = (1.0f - z) * nn + z * h;
            hlds[b * H2 + j] = hn;
            buf_outs[(size_t)r * HDIM + outOff + j] = hn;
        }
        __syncthreads();
    }
}

// ---------------------------------------------------------------------------
// K3: to_push (h0 == 0 so fully parallel): gates from GI[:,384:768] + cb_hh
__global__ void to_push_kernel(const float* __restrict__ GI, const float* __restrict__ cb_hh,
                               float* __restrict__ tp) {
    int idx = blockIdx.x * 256 + threadIdx.x;
    if (idx >= ROWS * HDIM) return;
    int r = idx >> 7, j = idx & 127;
    const float* g = GI + (size_t)r * 768 + 384;
    float rg = sigmoidf_(g[j]        + cb_hh[j]);
    float z  = sigmoidf_(g[128 + j]  + cb_hh[128 + j]);
    float nn = tanhf_(g[256 + j] + rg * cb_hh[256 + j]);
    tp[idx] = (1.0f - z) * nn;
}

// ---------------------------------------------------------------------------
// K4: attention. One block per (t,b), 128 threads.
__global__ __launch_bounds__(128) void attention(const float* __restrict__ bo,
                                                 const float* __restrict__ tp,
                                                 float* __restrict__ ba) {
    int t = blockIdx.x >> 5, b = blockIdx.x & 31;
    int tid = threadIdx.x;
    __shared__ float tpl[128];
    __shared__ float sc[128];
    __shared__ float red[128];
    tpl[tid] = tp[((size_t)(t * BSZ + b)) * HDIM + tid];
    __syncthreads();
    const float* row = bo + ((size_t)(tid * BSZ + b)) * HDIM;
    float acc = 0.0f;
    #pragma unroll 8
    for (int k = 0; k < HDIM; ++k) acc += row[k] * tpl[k];
    sc[tid] = acc; red[tid] = acc;
    __syncthreads();
    for (int o = 64; o > 0; o >>= 1) { if (tid < o) red[tid] = fmaxf(red[tid], red[tid + o]); __syncthreads(); }
    float mx = red[0];
    __syncthreads();
    float e = __expf(sc[tid] - mx);
    sc[tid] = e; red[tid] = e;
    __syncthreads();
    for (int o = 64; o > 0; o >>= 1) { if (tid < o) red[tid] += red[tid + o]; __syncthreads(); }
    float inv = fast_rcp(red[0]);
    __syncthreads();
    sc[tid] *= inv;
    __syncthreads();
    float acc2 = 0.0f;
    for (int s = 0; s < SEQ; ++s)
        acc2 += sc[s] * bo[((size_t)(s * BSZ + b)) * HDIM + tid];
    ba[((size_t)(t * BSZ + b)) * HDIM + tid] = acc2;
}

// ---------------------------------------------------------------------------
// K5: act_pre = to_push @ w_tp^T + buf_atten @ w_ba^T + act_b
__global__ void actpre_kernel(const float* __restrict__ tp, const float* __restrict__ ba,
                              const float* __restrict__ act_w, const float* __restrict__ act_b,
                              float* __restrict__ ap) {
    int idx = blockIdx.x * 256 + threadIdx.x;
    if (idx >= ROWS * NACT) return;
    int r = idx / NACT, a = idx - r * NACT;
    const float* w  = act_w + (size_t)a * 384;
    const float* x1 = tp + (size_t)r * HDIM;
    const float* x2 = ba + (size_t)r * HDIM;
    float acc = act_b[a];
    #pragma unroll 8
    for (int k = 0; k < HDIM; ++k) acc += x1[k] * w[k] + x2[k] * w[256 + k];
    ap[idx] = acc;
}

// ---------------------------------------------------------------------------
// K6: stack automaton scan. Batch elements are independent -> 32 blocks.
__global__ __launch_bounds__(128) void stack_scan(const float* __restrict__ ap,
                                                  const float* __restrict__ tp,
                                                  const float* __restrict__ act_w,
                                                  float* __restrict__ fstk) {
    int b = blockIdx.x, tid = threadIdx.x;   // 128 threads = one h column each
    __shared__ float stk[NSTK * HDIM];       // 8 KB
    __shared__ float action[NACT];
    __shared__ float ppS[NSTK], cS[NSTK];
    __shared__ float pushS, passS;
    for (int i = tid; i < NSTK * HDIM; i += 128) stk[i] = 0.0f;
    __syncthreads();
    for (int t = 0; t < SEQ; ++t) {
        if (tid < NACT) {
            float acc = ap[((size_t)(t * BSZ + b)) * NACT + tid];
            const float* w = act_w + (size_t)tid * 384 + 128;   // w_st row
            #pragma unroll 8
            for (int k = 0; k < HDIM; ++k) acc += stk[15 * HDIM + k] * w[k];
            action[tid] = acc;
        }
        __syncthreads();
        if (tid < NSTK) { float p = action[tid + 1]; ppS[tid] = p; cS[tid] = p * fast_rcp((float)(tid + 1)); }
        if (tid == 16) pushS = action[0];
        if (tid == 17) passS = action[NACT - 1];
        __syncthreads();
        float col[NSTK];
        #pragma unroll
        for (int j = 0; j < NSTK; ++j) col[j] = stk[j * HDIM + tid];
        float S = 0.0f, popavg = 0.0f;
        #pragma unroll
        for (int k = 1; k <= NSTK; ++k) { S += col[NSTK - k]; popavg += cS[k - 1] * S; }
        float tpv = tp[((size_t)(t * BSZ + b)) * HDIM + tid];
        float nw[NSTK];
        #pragma unroll
        for (int i = 0; i < NSTK; ++i) {
            float conv = 0.0f;
            for (int j = 0; j < i; ++j) conv += ppS[i - 1 - j] * col[j];
            float pushv = (i < NSTK - 1) ? col[i + 1] : tpv;
            nw[i] = conv + popavg + pushS * pushv + passS * col[i];
        }
        __syncthreads();
        #pragma unroll
        for (int i = 0; i < NSTK; ++i) stk[i * HDIM + tid] = nw[i];
        __syncthreads();
    }
    fstk[(size_t)b * HDIM + tid] = stk[15 * HDIM + tid];
}

// K7: neg_log_prob
__global__ void finalize(const float* __restrict__ fstk, const float* __restrict__ top_w,
                         const float* __restrict__ top_b, float* __restrict__ out) {
    __shared__ float red[BSZ];
    int tid = threadIdx.x;  // 32
    float acc = top_b[0];
    #pragma unroll 8
    for (int k = 0; k < HDIM; ++k) acc += fstk[(size_t)tid * HDIM + k] * top_w[k];
    float ls = fminf(acc, 0.0f) - log1pf(__expf(-fabsf(acc)));
    red[tid] = ls;
    __syncthreads();
    if (tid == 0) {
        float s = 0.0f;
        for (int i = 0; i < BSZ; ++i) s += red[i];
        out[NEG_OFF] = -(s / (float)BSZ);
    }
}

// ---------------------------------------------------------------------------
// K8: big logits GEMM. P = A(4096x128,f16) @ B^T (B = emb_w f16, row-major).
// Block = 128 threads (4 waves); wave tile = 64(M) x 32(N) = 4x2 WMMA accs.
// Row r of P writes to logits_right[r] (r<4064) and logits_left[r-32] (r>=32).
__global__ __launch_bounds__(128) void logits_gemm(const _Float16* __restrict__ A,
                                                   const _Float16* __restrict__ B,
                                                   float* __restrict__ out) {
    const int lane = threadIdx.x & 31;
    const int wave = threadIdx.x >> 5;            // 0..3
    const int row0 = blockIdx.x * 64;             // M base (4096/64 = 64)
    const int col0 = blockIdx.y * 128 + wave * 32;// N base (16000/128 = 125)
    const int lmod = lane & 15;
    const int lhi  = lane >> 4;

    float8 acc[4][2];
    #pragma unroll
    for (int ri = 0; ri < 4; ++ri)
        #pragma unroll
        for (int ci = 0; ci < 2; ++ci)
            acc[ri][ci] = (float8){0.f,0.f,0.f,0.f,0.f,0.f,0.f,0.f};

    #pragma unroll
    for (int k0 = 0; k0 < 128; k0 += 32) {
        half16 af[4];
        #pragma unroll
        for (int ri = 0; ri < 4; ++ri) {
            const _Float16* p = A + (size_t)(row0 + ri * 16 + lmod) * 128 + k0 + lhi * 8;
            union { half16 v; float4 f4[2]; } u;
            u.f4[0] = *(const float4*)(p);
            u.f4[1] = *(const float4*)(p + 16);
            af[ri] = u.v;
        }
        half16 bfr[2];
        #pragma unroll
        for (int ci = 0; ci < 2; ++ci) {
            const _Float16* p = B + (size_t)(col0 + ci * 16 + lmod) * 128 + k0 + lhi * 8;
            union { half16 v; float4 f4[2]; } u;
            u.f4[0] = *(const float4*)(p);
            u.f4[1] = *(const float4*)(p + 16);
            bfr[ci] = u.v;
        }
        #pragma unroll
        for (int ri = 0; ri < 4; ++ri)
            #pragma unroll
            for (int ci = 0; ci < 2; ++ci)
                acc[ri][ci] = __builtin_amdgcn_wmma_f32_16x16x32_f16(
                        false, af[ri], false, bfr[ci], (short)0, acc[ri][ci], false, false);
    }

    // Scatter to the two shifted logits tensors with streaming stores.
    #pragma unroll
    for (int ri = 0; ri < 4; ++ri) {
        #pragma unroll
        for (int ci = 0; ci < 2; ++ci) {
            int gc  = col0 + ci * 16 + lmod;
            int grb = row0 + ri * 16 + (lhi << 3);
            #pragma unroll
            for (int v = 0; v < 8; ++v) {
                int gr = grb + v;
                float val = acc[ri][ci][v];
                if (gr >= BSZ)                               // logits_left[s-1]
                    __builtin_nontemporal_store(val, out + (size_t)(gr - BSZ) * VOC + gc);
                if (gr < LOGROWS)                            // logits_right[s]
                    __builtin_nontemporal_store(val, out + RIGHT_OFF + (size_t)gr * VOC + gc);
            }
        }
    }
}

// ---------------------------------------------------------------------------
extern "C" void kernel_launch(void* const* d_in, const int* in_sizes, int n_in,
                              void* d_out, int out_size, void* d_ws, size_t ws_size,
                              hipStream_t stream) {
    const int*   inputs = (const int*)  d_in[0];
    const float* emb_w  = (const float*)d_in[1];
    const float* bwf_ih = (const float*)d_in[2];
    const float* bwf_hh = (const float*)d_in[3];
    const float* bbf_ih = (const float*)d_in[4];
    const float* bbf_hh = (const float*)d_in[5];
    const float* bwb_ih = (const float*)d_in[6];
    const float* bwb_hh = (const float*)d_in[7];
    const float* bbb_ih = (const float*)d_in[8];
    const float* bbb_hh = (const float*)d_in[9];
    const float* cw_ih  = (const float*)d_in[10];
    const float* cw_hh  = (const float*)d_in[11]; (void)cw_hh; // h0==0: only bias matters
    const float* cb_ih  = (const float*)d_in[12];
    const float* cb_hh  = (const float*)d_in[13];
    const float* act_w  = (const float*)d_in[14];
    const float* act_b  = (const float*)d_in[15];
    const float* top_w  = (const float*)d_in[16];
    const float* top_b  = (const float*)d_in[17];
    float* out = (float*)d_out;

    // Workspace layout (floats)
    float* ws   = (float*)d_ws;
    float* embs = ws;                          // 4096*128   = 524288
    float* GI   = embs + ROWS * EDIM;          // 4096*768   = 3145728
    float* tp   = GI   + ROWS * 768;           // 524288
    float* bo   = tp   + ROWS * HDIM;          // 524288
    float* ba   = bo   + ROWS * HDIM;          // 524288
    float* ap   = ba   + ROWS * HDIM;          // 73728
    float* fstk = ap   + ROWS * NACT;          // 4096
    _Float16* a_h   = (_Float16*)(fstk + BSZ * HDIM);   // 4096*128 halfs
    _Float16* emb_h = (_Float16*)((float*)a_h + (ROWS * HDIM) / 2); // 16000*128 halfs

    embed_gather<<<ROWS, EDIM, 0, stream>>>(inputs, emb_w, embs);
    cast_f16<<<(VOC * EDIM + 255) / 256, 256, 0, stream>>>(emb_w, emb_h, VOC * EDIM);
    gi_precompute<<<(ROWS * 768) / 256, 256, 0, stream>>>(
        embs, bwf_ih, bbf_ih, bwb_ih, bbb_ih, cw_ih, cb_ih, GI);
    gru_scan<<<2, 256, 0, stream>>>(GI, bwf_hh, bbf_hh, bwb_hh, bbb_hh, bo);
    to_push_kernel<<<(ROWS * HDIM) / 256, 256, 0, stream>>>(GI, cb_hh, tp);
    cast_f16<<<(ROWS * HDIM + 255) / 256, 256, 0, stream>>>(bo, a_h, ROWS * HDIM);
    attention<<<ROWS, 128, 0, stream>>>(bo, tp, ba);
    actpre_kernel<<<(ROWS * NACT + 255) / 256, 256, 0, stream>>>(tp, ba, act_w, act_b, ap);
    logits_gemm<<<dim3(ROWS / 64, VOC / 128), 128, 0, stream>>>(a_h, emb_h, out);
    stack_scan<<<BSZ, 128, 0, stream>>>(ap, tp, act_w, fstk);
    finalize<<<1, BSZ, 0, stream>>>(fstk, top_w, top_b, out);
}
